// SimpleNet_61710090108974
// MI455X (gfx1250) — compile-verified, matrix operand order checked by reading, not run on hardware
//
#include <hip/hip_runtime.h>

// SimpleNet per-pixel MLP on CDNA5 WMMA (gfx1250, wave32).
// Layer 0: V_WMMA_F32_16X16X4_F32 (K=4 >= C_IN=3) straight from f32 x loads.
// Layers 1..7: [32x32] f16 A-tiles x [32 x 16px] f16 B, f32 accum.
// f32 C/D layout == f16 B-operand layout (M->K), so inter-layer repack is
// per-lane v_tanh_f32 + v_cvt_pk_f16_f32 (no LDS/shuffles in the chain).
// Weights staged once per block into a padded f16 LDS image; each 16x32
// A-fragment is then two ds_load_b128 (contiguous 8-half K-runs per row).

typedef __attribute__((ext_vector_type(16))) _Float16 v16h;
typedef __attribute__((ext_vector_type(8)))  _Float16 v8h;
typedef __attribute__((ext_vector_type(8)))  float    v8f;
typedef __attribute__((ext_vector_type(2)))  float    v2f;

#define HW_N    (1024 * 1024)
#define BATCH   4
#define CIN     3
#define MAXH    25
#define NPIX    (BATCH * HW_N)
#define NCHAIN  4
#define NGROUPS (NPIX / (16 * NCHAIN))   // 64 pixels per loop iteration

#if __has_builtin(__builtin_amdgcn_wmma_f32_16x16x4_f32)
#define HAVE_K4F32 1
#else
#define HAVE_K4F32 0
#endif

__device__ __forceinline__ float fast_tanh(float x) {
#if __has_builtin(__builtin_amdgcn_tanhf)
  return __builtin_amdgcn_tanhf(x);
#elif __has_builtin(__builtin_amdgcn_tanh_f32)
  return __builtin_amdgcn_tanh_f32(x);
#else
  return tanhf(x);
#endif
}

__device__ __forceinline__ float fast_sigmoid(float x) {
#if __has_builtin(__builtin_amdgcn_exp2f) && __has_builtin(__builtin_amdgcn_rcpf)
  return __builtin_amdgcn_rcpf(1.0f + __builtin_amdgcn_exp2f(x * -1.44269504088896341f));
#else
  return 1.0f / (1.0f + expf(-x));
#endif
}

// Padded weight element for logical layer l (0=in, 1..6=mid, 7=out).
__device__ __forceinline__ float wload(const float* __restrict__ w_in,
                                       const float* __restrict__ ws,
                                       const float* __restrict__ w_out,
                                       int l, int m, int k) {
  if (l == 0) return (m < MAXH && k < CIN) ? w_in[m * CIN + k] : 0.0f;
  if (l == 7) return (m < CIN && k < MAXH) ? w_out[m * MAXH + k] : 0.0f;
  return (m < MAXH && k < MAXH) ? ws[(l - 1) * MAXH * MAXH + m * MAXH + k] : 0.0f;
}

__global__ __launch_bounds__(256, 1) void simplenet_wmma_kernel(
    const float* __restrict__ x, const float* __restrict__ w_in,
    const float* __restrict__ ws, const float* __restrict__ w_out,
    float* __restrict__ out) {
  // Padded f16 weight image: 8 layers x 32 rows x 32 cols = 16 KB of LDS.
  __shared__ __align__(16) _Float16 wlds[8 * 32 * 32];

  const int tid  = threadIdx.x;
  const int lane = tid & 31;
  const int lo   = lane & 15;   // N (pixel) / M row within tile
  const int hi   = lane >> 4;   // selects K-halves per A/B layout
  const int wave   = (blockIdx.x * (blockDim.x >> 5)) + (tid >> 5);
  const int nwaves = gridDim.x * (blockDim.x >> 5);

  // ---- Stage weights: thread (grp,kk) covers column kk of rows grp*32+j.
  // Within a wave all lanes share a row and read consecutive k: coalesced.
  {
    const int kk  = tid & 31;
    const int grp = tid >> 5;
    for (int j = 0; j < 32; ++j) {
      const int row = grp * 32 + j;       // 0..255 = (layer<<5)|m
      wlds[row * 32 + kk] = (_Float16)wload(w_in, ws, w_out, row >> 5, row & 31, kk);
    }
  }
  __syncthreads();

#if HAVE_K4F32
  // Layer-0 A as f32 16x4 tiles: lane = M; VGPR j holds K = 2*hi + j.
  v2f A0[2];
#pragma unroll
  for (int t = 0; t < 2; ++t) {
    const int m = t * 16 + lo;
#pragma unroll
    for (int j = 0; j < 2; ++j) {
      const int k = 2 * hi + j;
      A0[t][j] = (m < MAXH && k < CIN) ? w_in[m * CIN + k] : 0.0f;
    }
  }
  // Per-lane selector: 1.0 for lanes<16 (keep c1), 0.0 for lanes>=16.
  // Used multiplicatively so the c1 load stays unconditional (no exec branch).
  const float hsel = (hi == 0) ? 1.0f : 0.0f;
#endif

  // ---- f16 A-fragments from LDS: 2 x ds_load_b128 each.
  // A layout (16-bit, 16x32): lane = M (mod 16); element e holds
  // K = e + (e & 8) + 8 * (lane >= 16)  ->  runs [8hi..8hi+7], [16+8hi..+7].
#if HAVE_K4F32
  v16h A[13];     // layers 1..6 x 2 tiles + out tile
  const int lfirst = 1;
#else
  v16h A[15];     // layers 0..6 x 2 tiles + out tile
  const int lfirst = 0;
#endif
#pragma unroll
  for (int l = lfirst; l < 7; ++l) {
#pragma unroll
    for (int t = 0; t < 2; ++t) {
      const _Float16* base = &wlds[((l * 32) + t * 16 + lo) * 32];
      const v8h a0 = *(const v8h*)(base + hi * 8);
      const v8h a1 = *(const v8h*)(base + 16 + hi * 8);
      A[(l - lfirst) * 2 + t] = __builtin_shufflevector(
          a0, a1, 0, 1, 2, 3, 4, 5, 6, 7, 8, 9, 10, 11, 12, 13, 14, 15);
    }
  }
  const int AOUT = (7 - lfirst) * 2;
  {
    const _Float16* base = &wlds[(7 * 32 + lo) * 32];
    const v8h a0 = *(const v8h*)(base + hi * 8);
    const v8h a1 = *(const v8h*)(base + 16 + hi * 8);
    A[AOUT] = __builtin_shufflevector(a0, a1, 0, 1, 2, 3, 4, 5, 6, 7,
                                      8, 9, 10, 11, 12, 13, 14, 15);
  }

  const v8f zc = {};

  for (int g = wave; g < NGROUPS; g += nwaves) {
    const int pix = g * (16 * NCHAIN);    // 64 consecutive pixels; HW_N % 64
    const int b   = pix >> 20;            // == 0 -> never crosses a batch.
    const int n   = pix & (HW_N - 1);
    const float* xb = x + (size_t)b * CIN * HW_N + n;

    v16h B[NCHAIN];
#if HAVE_K4F32
    // Layer 0 in f32: B0 VGPR0 = x[c0] (lanes<16) | x[c2] (lanes>=16),
    // VGPR1 = x[c1] * hsel.  Fully branchless; all lanes issue both loads.
#pragma unroll
    for (int q = 0; q < NCHAIN; ++q) {
      const float* xa = xb + q * 16 + lo;
      const float b0  = xa[(size_t)hi * (2 * HW_N)];  // c0 or c2
      const float xc1 = xa[HW_N];                     // unconditional load
      const float b1  = xc1 * hsel;                   // c1 or 0, via v_mul
      const v2f Bv = {b0, b1};
      const v8f e0 = __builtin_amdgcn_wmma_f32_16x16x4_f32(false, A0[0], false, Bv, (short)0, zc, false, false);
      const v8f e1 = __builtin_amdgcn_wmma_f32_16x16x4_f32(false, A0[1], false, Bv, (short)0, zc, false, false);
      v16h nb;
#pragma unroll
      for (int r = 0; r < 8; ++r) {
        nb[r]     = (_Float16)fast_tanh(e0[r]);
        nb[r + 8] = (_Float16)fast_tanh(e1[r]);
      }
      B[q] = nb;
    }
#else
    // Fallback: layer 0 via f16 WMMA with zero-padded B.
#pragma unroll
    for (int q = 0; q < NCHAIN; ++q) B[q] = (v16h){};
    if (lane < 16) {
#pragma unroll
      for (int q = 0; q < NCHAIN; ++q) {
        const float* xa = xb + q * 16 + lo;
        B[q][0] = (_Float16)xa[0];
        B[q][1] = (_Float16)xa[HW_N];
        B[q][2] = (_Float16)xa[2 * HW_N];
      }
    }
#pragma unroll
    for (int q = 0; q < NCHAIN; ++q) {
      const v8f e0 = __builtin_amdgcn_wmma_f32_16x16x32_f16(false, A[0], false, B[q], (short)0, zc, false, false);
      const v8f e1 = __builtin_amdgcn_wmma_f32_16x16x32_f16(false, A[1], false, B[q], (short)0, zc, false, false);
      v16h nb;
#pragma unroll
      for (int r = 0; r < 8; ++r) {
        nb[r]     = (_Float16)fast_tanh(e0[r]);
        nb[r + 8] = (_Float16)fast_tanh(e1[r]);
      }
      B[q] = nb;
    }
#endif

    // Layers 1..6 in f16; 4 independent chains per layer.
#pragma unroll
    for (int l = 1; l < 7; ++l) {
      const int ai = (l - lfirst) * 2;
      v8f d0[NCHAIN], d1[NCHAIN];
#pragma unroll
      for (int q = 0; q < NCHAIN; ++q)
        d0[q] = __builtin_amdgcn_wmma_f32_16x16x32_f16(false, A[ai + 0], false, B[q], (short)0, zc, false, false);
#pragma unroll
      for (int q = 0; q < NCHAIN; ++q)
        d1[q] = __builtin_amdgcn_wmma_f32_16x16x32_f16(false, A[ai + 1], false, B[q], (short)0, zc, false, false);
#pragma unroll
      for (int q = 0; q < NCHAIN; ++q) {
        v16h nb;
#pragma unroll
        for (int r = 0; r < 8; ++r) {
          nb[r]     = (_Float16)fast_tanh(d0[q][r]);
          nb[r + 8] = (_Float16)fast_tanh(d1[q][r]);
        }
        B[q] = nb;
      }
    }

    // Output layer: only 3 rows live -> M-tile 0 only.
    v8f o[NCHAIN];
#pragma unroll
    for (int q = 0; q < NCHAIN; ++q)
      o[q] = __builtin_amdgcn_wmma_f32_16x16x32_f16(false, A[AOUT], false, B[q], (short)0, zc, false, false);

    // D rows 0..2 (= out channels) live in VGPRs 0..2 of lanes 0..15.
    if (lane < 16) {
      float* op = out + (size_t)b * CIN * HW_N + n;
#pragma unroll
      for (int c = 0; c < CIN; ++c) {
#pragma unroll
        for (int q = 0; q < NCHAIN; ++q)
          op[c * HW_N + q * 16 + lo] = fast_sigmoid(o[q][c]);
      }
    }
  }
}

extern "C" void kernel_launch(void* const* d_in, const int* in_sizes, int n_in,
                              void* d_out, int out_size, void* d_ws, size_t ws_size,
                              hipStream_t stream) {
  (void)in_sizes; (void)n_in; (void)d_ws; (void)ws_size; (void)out_size;
  const float* x     = (const float*)d_in[0];
  const float* w_in  = (const float*)d_in[1];
  const float* ws    = (const float*)d_in[2];
  const float* w_out = (const float*)d_in[3];
  float* out = (float*)d_out;

  // 2048 blocks x 8 waves = 16384 waves; 65536 pixel-groups -> 4 iters/wave.
  dim3 grid(2048), block(256);
  simplenet_wmma_kernel<<<grid, block, 0, stream>>>(x, w_in, ws, w_out, out);
}
